// LogicConv3d_69415261438611
// MI455X (gfx1250) — matrix operand order repaired and checked by reading, not compiled, not run on hardware
//
#include <hip/hip_runtime.h>

// ---------------------------------------------------------------------------
// LogicConv3d for MI455X (gfx1250, wave32).
//  Kernel B (first): coefficient einsum  softmax(w) @ GATE_M  on the WMMA unit.
//  Kernel A (LAST in file so disasm snippet shows its metadata): LDS-staged
//            gather (async-to-LDS) + register binary-tree fold, unroll-by-8.
// ---------------------------------------------------------------------------

typedef __attribute__((ext_vector_type(16))) _Float16 v16h;
typedef __attribute__((ext_vector_type(8)))  float    v8f;

#define Bn   16
#define Cn   64
#define Hn   32
#define Wn   32
#define Kn   64
#define NPos 30
#define Pn   900      // 30*30
#define Sn   64
#define NROWS 8128    // (64+32+16+8+4+2+1)*64  total (lvl,s,k) gates

__device__ __constant__ float GATE_M_d[16][4] = {
    {0,0,0,0},{0,0,0,1},{0,1,0,-1},{0,1,0,0},
    {0,0,1,-1},{0,0,1,0},{0,1,1,-2},{0,1,1,-1},
    {1,-1,-1,1},{1,-1,-1,2},{1,0,-1,0},{1,0,-1,1},
    {1,-1,0,0},{1,-1,0,1},{1,0,0,-1},{1,0,0,0}};

// gate node: c.x + c.y*a + c.z*b + c.w*a*b, componentwise over 4 outputs
__device__ __forceinline__ float4 mix4(float4 a, float4 b, float4 c) {
    float4 r;
    r.x = c.x + c.y * a.x + c.z * b.x + c.w * (a.x * b.x);
    r.y = c.x + c.y * a.y + c.z * b.y + c.w * (a.y * b.y);
    r.z = c.x + c.y * a.z + c.z * b.z + c.w * (a.z * b.z);
    r.w = c.x + c.y * a.w + c.z * b.w + c.w * (a.w * b.w);
    return r;
}

// -------------------------------------------------------------------------
// Kernel B: one wave computes 16 rows of  coef[row][0:4] = softmax(w_row) @ G
// Rows are flattened over levels: global row = GOFF[lvl] + s*64 + k.
// WMMA: A = probs (16x32 f16, K 16..31 zero), B = GATE_M padded (32x16 f16).
// -------------------------------------------------------------------------
__global__ __launch_bounds__(32)
void coef_wmma_kernel(const float* __restrict__ w0, const float* __restrict__ w1,
                      const float* __restrict__ w2, const float* __restrict__ w3,
                      const float* __restrict__ w4, const float* __restrict__ w5,
                      const float* __restrict__ w6, float* __restrict__ coefOut)
{
    const int lane = threadIdx.x;            // 0..31
    const int base = blockIdx.x * 16;        // 508 blocks * 16 rows = 8128
    const int M    = lane & 15;
    const bool hi  = lane >= 16;
    const int r    = base + M;               // my A-matrix row (global gate id)

    const int lvl  = (r < 4096) ? 0 : (r < 6144) ? 1 : (r < 7168) ? 2 :
                     (r < 7680) ? 3 : (r < 7936) ? 4 : (r < 8064) ? 5 : 6;
    const int goff = (lvl == 0) ? 0 : (lvl == 1) ? 4096 : (lvl == 2) ? 6144 :
                     (lvl == 3) ? 7168 : (lvl == 4) ? 7680 : (lvl == 5) ? 7936 : 8064;
    const float* wp;
    switch (lvl) {
        case 0: wp = w0; break; case 1: wp = w1; break; case 2: wp = w2; break;
        case 3: wp = w3; break; case 4: wp = w4; break; case 5: wp = w5; break;
        default: wp = w6; break;
    }
    // 16 logits of my row, 64B-aligned -> 4x b128
    const float4* lg4 = (const float4*)(wp + (size_t)(r - goff) * 16);
    float l[16];
    {
        float4 q0 = lg4[0], q1 = lg4[1], q2 = lg4[2], q3 = lg4[3];
        l[0]=q0.x; l[1]=q0.y; l[2]=q0.z; l[3]=q0.w;
        l[4]=q1.x; l[5]=q1.y; l[6]=q1.z; l[7]=q1.w;
        l[8]=q2.x; l[9]=q2.y; l[10]=q2.z; l[11]=q2.w;
        l[12]=q3.x; l[13]=q3.y; l[14]=q3.z; l[15]=q3.w;
    }
    float mx = -3.0e38f;
#pragma unroll
    for (int g = 0; g < 16; ++g) mx = fmaxf(mx, l[g]);
    float sum = 0.f, pr[16];
#pragma unroll
    for (int g = 0; g < 16; ++g) { pr[g] = __expf(l[g] - mx); sum += pr[g]; }
    const float inv = 1.f / sum;

    // A fragment (16-bit A 16x32): lanes 0-15 hold K=0..7 then K=16..23,
    // lanes 16-31 hold K=8..15 then K=24..31. K>=16 is zero padding.
    v16h a;
#pragma unroll
    for (int e = 0; e < 8; ++e) a[e] = (_Float16)(pr[(hi ? 8 : 0) + e] * inv);
#pragma unroll
    for (int e = 8; e < 16; ++e) a[e] = (_Float16)0.f;

    // B fragment (32x16): lanes 0-15 hold K'=0..7 then K'=16..23,
    // lanes 16-31 hold K'=8..15 then K'=24..31 (GATE_M rows, cols<4; K'>=16 zero).
    v16h bf;
    const int n = lane & 15;
#pragma unroll
    for (int e = 0; e < 16; ++e) {
        float g = (e < 8 && n < 4) ? GATE_M_d[(hi ? 8 : 0) + e][n] : 0.f;
        bf[e] = (_Float16)g;
    }

    v8f c = {};
    c = __builtin_amdgcn_wmma_f32_16x16x32_f16(false, a, false, bf,
                                               (short)0, c, false, false);

    // D layout: VGPR r2 -> row (hi ? r2+8 : r2), col = lane&15.
    if (n < 4) {
#pragma unroll
        for (int r2 = 0; r2 < 8; ++r2) {
            int row = base + r2 + (hi ? 8 : 0);
            coefOut[row * 4 + n] = c[r2];
        }
    }
}

// -------------------------------------------------------------------------
// Main kernel: per block: b fixed, stage x[b] (256KB) into LDS via async
// copies; each wave owns k = kg*8 + wave and 128 output positions.
// -------------------------------------------------------------------------
#define XS_FLOATS (Cn * Hn * Wn)           // 65536 floats = 256 KB
#define IDX_INTS  (8 * Sn * 2)             // precomputed abase/bbase per (wave,s)
#define CF_FLOATS (8 * 127 * 4)            // 4064
#define SMEM_BYTES ((XS_FLOATS + IDX_INTS + CF_FLOATS) * 4)   // 282,496 B

__global__ __launch_bounds__(256, 1)       // 1 block/WGP (LDS-bound)
void logicconv_main(const float* __restrict__ x,
                    const int*   __restrict__ idxA,
                    const int*   __restrict__ idxB,
                    const float* __restrict__ coef,
                    float*       __restrict__ out)
{
    extern __shared__ float smem[];
    float* xs   = smem;                                   // x[b] slab
    int*   idxs = (int*)(smem + XS_FLOATS);               // gather bases per wave-k
    float* cfs  = smem + XS_FLOATS + IDX_INTS;            // coefs per wave-k

    const int tid  = threadIdx.x;
    const int lane = tid & 31;
    const int wv   = tid >> 5;
    const int ptile = blockIdx.x;        // 0..7  (128 p each)
    const int kg    = blockIdx.y;        // 0..7
    const int b     = blockIdx.z;        // 0..15
    const int k     = kg * 8 + wv;

    // ---- stage x[b] -> LDS with CDNA5 async-to-LDS (ASYNCcnt) ----
    const float* xb = x + (size_t)b * XS_FLOATS;
    for (int i = tid; i < XS_FLOATS / 4; i += 256) {
        unsigned voff  = (unsigned)(i * 16);                           // byte off
        unsigned laddr = (unsigned)(size_t)(const void*)(xs + i * 4);  // LDS addr
        asm volatile("global_load_async_to_lds_b128 %0, %1, %2"
                     :: "v"(laddr), "v"(voff), "s"(xb) : "memory");
    }

    // ---- gather bases from the p==0 slice of idx tensors (separable) ----
    for (int s = lane; s < Sn; s += 32) {
        const int* pa = idxA + ((size_t)k * (Pn * Sn) + s) * 3;   // (k, p=0, s, :)
        const int* pb = idxB + ((size_t)k * (Pn * Sn) + s) * 3;
        int2 v;
        v.x = pa[2] * (Hn * Wn) + pa[0] * Wn + pa[1];   // abase = c*1024 + h*32 + w
        v.y = pb[2] * (Hn * Wn) + pb[0] * Wn + pb[1];
        *(int2*)&idxs[wv * (Sn * 2) + s * 2] = v;
    }

    // ---- coefficients for my k (127 gates * 4 coefs) ----
    for (int r = lane; r < 127; r += 32) {
        int lvl  = (r < 64) ? 0 : (r < 96) ? 1 : (r < 112) ? 2 :
                   (r < 120) ? 3 : (r < 124) ? 4 : (r < 126) ? 5 : 6;
        int loff = (lvl == 0) ? 0 : (128 - (128 >> lvl));
        int goff = (lvl == 0) ? 0 : (lvl == 1) ? 4096 : (lvl == 2) ? 6144 :
                   (lvl == 3) ? 7168 : (lvl == 4) ? 7680 : (lvl == 5) ? 7936 : 8064;
        int grow = goff + (r - loff) * Kn + k;
        float4 v = *(const float4*)(coef + (size_t)grow * 4);
        *(float4*)&cfs[wv * (127 * 4) + r * 4] = v;
    }

    asm volatile("s_wait_asynccnt 0" ::: "memory");
    __syncthreads();

    // ---- per-lane: 4 output positions (components of int4) ----
    int4 pOff; unsigned actMask = 0;
    {
        int p0 = ptile * 128 + lane;
        int p1 = p0 + 32, p2 = p0 + 64, p3 = p0 + 96;
        int h0 = p0 / NPos, h1 = p1 / NPos, h2 = p2 / NPos, h3 = p3 / NPos;
        pOff.x = (p0 < Pn) ? (h0 * Wn + (p0 - h0 * NPos)) : 0;
        pOff.y = (p1 < Pn) ? (h1 * Wn + (p1 - h1 * NPos)) : 0;
        pOff.z = (p2 < Pn) ? (h2 * Wn + (p2 - h2 * NPos)) : 0;
        pOff.w = (p3 < Pn) ? (h3 * Wn + (p3 - h3 * NPos)) : 0;
        actMask = (p0 < Pn) | ((p1 < Pn) << 1) | ((p2 < Pn) << 2) | ((p3 < Pn) << 3);
    }

    const int*   myIdx = &idxs[wv * (Sn * 2)];
    const float* myCf  = &cfs[wv * (127 * 4)];

    // fold stack: 7 slots, ALL indices are literal constants -> registers.
    float4 st[7];

    // unroll-by-8: levels 1-3 fold statically inside each group; levels 4-6
    // are rare wave-uniform branches. Keeps the body small -> no RA spills.
#pragma unroll 8
    for (int s = 0; s < Sn; ++s) {
        int2   ab = *(const int2*)&myIdx[s * 2];          // ds_load_b64
        float4 c0 = *(const float4*)&myCf[s * 4];         // ds_load_b128
        float4 av, bv;
        av.x = xs[ab.x + pOff.x];  bv.x = xs[ab.y + pOff.x];
        av.y = xs[ab.x + pOff.y];  bv.y = xs[ab.y + pOff.y];
        av.z = xs[ab.x + pOff.z];  bv.z = xs[ab.y + pOff.z];
        av.w = xs[ab.x + pOff.w];  bv.w = xs[ab.y + pOff.w];
        float4 v = mix4(av, bv, c0);

        if ((s & 1) == 1)
            v = mix4(st[0], v, *(const float4*)&myCf[(64  + (s >> 1)) * 4]);
        if ((s & 3) == 3)
            v = mix4(st[1], v, *(const float4*)&myCf[(96  + (s >> 2)) * 4]);
        if ((s & 7) == 7)
            v = mix4(st[2], v, *(const float4*)&myCf[(112 + (s >> 3)) * 4]);
        if ((s & 15) == 15)
            v = mix4(st[3], v, *(const float4*)&myCf[(120 + (s >> 4)) * 4]);
        if ((s & 31) == 31)
            v = mix4(st[4], v, *(const float4*)&myCf[(124 + (s >> 5)) * 4]);
        if (s == 63)
            v = mix4(st[5], v, *(const float4*)&myCf[126 * 4]);

        if      ((s & 1)  == 0) st[0] = v;
        else if ((s & 2)  == 0) st[1] = v;
        else if ((s & 4)  == 0) st[2] = v;
        else if ((s & 8)  == 0) st[3] = v;
        else if ((s & 16) == 0) st[4] = v;
        else if ((s & 32) == 0) st[5] = v;
        else                    st[6] = v;
    }

    float* op = out + ((size_t)b * Kn + k) * Pn + (ptile * 128 + lane);
    if (actMask & 1) op[0]  = st[6].x;
    if (actMask & 2) op[32] = st[6].y;
    if (actMask & 4) op[64] = st[6].z;
    if (actMask & 8) op[96] = st[6].w;
}

// -------------------------------------------------------------------------
extern "C" void kernel_launch(void* const* d_in, const int* in_sizes, int n_in,
                              void* d_out, int out_size, void* d_ws, size_t ws_size,
                              hipStream_t stream) {
    const float* x    = (const float*)d_in[0];
    const int*   idxA = (const int*)  d_in[1];
    const int*   idxB = (const int*)  d_in[2];
    const float* w0   = (const float*)d_in[3];
    const float* w1   = (const float*)d_in[4];
    const float* w2   = (const float*)d_in[5];
    const float* w3   = (const float*)d_in[6];
    const float* w4   = (const float*)d_in[7];
    const float* w5   = (const float*)d_in[8];
    const float* w6   = (const float*)d_in[9];
    float* coef = (float*)d_ws;          // NROWS*4 floats = 130 KB scratch
    float* outp = (float*)d_out;

    coef_wmma_kernel<<<dim3(NROWS / 16), dim3(32), 0, stream>>>(
        w0, w1, w2, w3, w4, w5, w6, coef);

    logicconv_main<<<dim3(8, 8, Bn), dim3(256), SMEM_BYTES, stream>>>(
        x, idxA, idxB, coef, outp);
}
